// Attention_47588237640034
// MI455X (gfx1250) — matrix operand order repaired and checked by reading, not compiled, not run on hardware
//
#include <hip/hip_runtime.h>
#include <hip/hip_bf16.h>

// CDNA5 / gfx1250 WMMA types
typedef __attribute__((ext_vector_type(16))) __bf16 v16bf;
typedef __attribute__((ext_vector_type(8)))  float  v8f;

union FragBF {
  v16bf v;
  __bf16 h[16];
  uint4  q[2];
};

__device__ inline v8f wmma_bf16(v16bf a, v16bf b, v8f c) {
  return __builtin_amdgcn_wmma_f32_16x16x32_bf16(false, a, false, b, (short)0, c, false, false);
}

__device__ inline v8f zero_v8f() {
  v8f z;
#pragma unroll
  for (int i = 0; i < 8; ++i) z[i] = 0.0f;
  return z;
}

// ---- CDNA5 async copy: global -> LDS, tracked by ASYNCcnt ----
__device__ inline void async_copy_b128(unsigned lds_off, const __bf16* gptr) {
  asm volatile("global_load_async_to_lds_b128 %0, %1, off"
               :: "v"(lds_off), "v"((unsigned long long)gptr) : "memory");
}
__device__ inline void wait_async_all() {
  asm volatile("s_wait_asynccnt 0x0" ::: "memory");
}
// generic __shared__ pointer -> 32-bit LDS byte address (addr[31:0] per ISA aperture rule)
__device__ inline unsigned lds_addr(const void* p) {
  return (unsigned)(unsigned long long)p;
}

// ---- bf16 fragment loaders: contiguous 16B runs per ISA 7.12.2 (wave32) ----
// A (16x32 MxK) from row-major bf16, K contiguous.
__device__ inline v16bf load_a_bf16(const __bf16* src, int ld) {
  const int lane = threadIdx.x & 31;
  const int m = lane & 15;
  const int kg = (lane >> 4) & 1;
  const __bf16* p = src + (long)m * ld + kg * 8;
  FragBF u;
  u.q[0] = *(const uint4*)p;
  u.q[1] = *(const uint4*)(p + 16);
  return u.v;
}

// B (32x16 KxN) where memory holds B^T row-major (row = output col n, K contiguous).
__device__ inline v16bf load_b_bf16t(const __bf16* src, int ld) {
  const int lane = threadIdx.x & 31;
  const int n = lane & 15;
  const int kb = (lane >> 4) << 3;
  const __bf16* p = src + (long)n * ld + kb;
  FragBF u;
  u.q[0] = *(const uint4*)p;
  u.q[1] = *(const uint4*)(p + 16);
  return u.v;
}

// cross-lane reductions within 16-lane halves
__device__ inline float hsum16(float v) {
#pragma unroll
  for (int m = 1; m <= 8; m <<= 1) v += __shfl_xor(v, m, 32);
  return v;
}
__device__ inline float hmax16(float v) {
#pragma unroll
  for (int m = 1; m <= 8; m <<= 1) v = fmaxf(v, __shfl_xor(v, m, 32));
  return v;
}

// ---------------------------------------------------------------------------
// Pre-pass: fp32 -> bf16 cast (8 elems/thread)
// ---------------------------------------------------------------------------
__global__ __launch_bounds__(256) void cast_bf16_kernel(
    const float* __restrict__ src, __bf16* __restrict__ dst) {
  const long i = ((long)blockIdx.x * 256 + threadIdx.x) * 8;
  const float4 a = *(const float4*)(src + i);
  const float4 b = *(const float4*)(src + i + 4);
  FragBF u;
  u.h[0] = (__bf16)a.x; u.h[1] = (__bf16)a.y;
  u.h[2] = (__bf16)a.z; u.h[3] = (__bf16)a.w;
  u.h[4] = (__bf16)b.x; u.h[5] = (__bf16)b.y;
  u.h[6] = (__bf16)b.z; u.h[7] = (__bf16)b.w;
  *(uint4*)(dst + i) = u.q[0];
}

// Pre-pass: transpose + cast: dst[c*R + r] = src[r*C + c]  (W -> W^T bf16)
__global__ __launch_bounds__(256) void transpose_cast_kernel(
    const float* __restrict__ src, __bf16* __restrict__ dst, int R, int C) {
  const long i = (long)blockIdx.x * 256 + threadIdx.x;
  if (i < (long)R * C) {
    const int c = (int)(i / R);
    const int r = (int)(i % R);
    dst[i] = (__bf16)src[(long)r * C + c];
  }
}

// ---------------------------------------------------------------------------
// Kernel 1: qkv = x @ w_qkv + b ; fused L2-normalize for q (x scale) and k.
// Grid (256, 24): 128-token row tile x (t,h) 64-col slice. Wave = 32x64.
// B^T tile (64x32, 4KB) double-buffered in LDS via async copies.
// q,k stored transposed [b][h][d][4096]; v natural [b][h][n][d].
// ---------------------------------------------------------------------------
__global__ __launch_bounds__(128) void qkv_norm_kernel(
    const __bf16* __restrict__ xb, const __bf16* __restrict__ wT,
    const float* __restrict__ bias,
    __bf16* __restrict__ qT, __bf16* __restrict__ kT, __bf16* __restrict__ vo) {
  __shared__ __align__(16) __bf16 bsm[2][64][32];  // B^T staging (double buffer)
  __shared__ __align__(16) __bf16 tsm[4][64][40];  // per-wave transpose [d][n+pad]
  const int tid  = threadIdx.x;
  const int lane = tid & 31;
  const int wave = tid >> 5;
  const int rt = blockIdx.x;            // 128-token row tile
  const int s  = blockIdx.y;            // 0..23
  const int t  = s >> 3;                // 0:q 1:k 2:v
  const int h  = s & 7;
  const int row0 = rt * 128 + wave * 32;
  const int b = row0 >> 12;
  const int nloc = row0 & 4095;
  const int bh = b * 8 + h;
  const int colbase = t * 512 + h * 64;

  v8f acc[2][4];
#pragma unroll
  for (int ra = 0; ra < 2; ++ra)
#pragma unroll
    for (int c = 0; c < 4; ++c) acc[ra][c] = zero_v8f();

  const __bf16* arow = xb + (long)row0 * 512;
  // stage helper: copy 64 rows x 32 K bf16 (256 x 16B chunks / 128 threads)
  const int S = 16;
  {
#pragma unroll
    for (int i = 0; i < 2; ++i) {
      const int ch = tid + i * 128;
      const int r = ch >> 2, part = ch & 3;
      async_copy_b128(lds_addr(&bsm[0][r][part * 8]),
                      wT + (long)(colbase + r) * 512 + part * 8);
    }
    wait_async_all();
    __syncthreads();
  }
  for (int st = 0; st < S; ++st) {
    const int kk = st * 32;
    if (st + 1 < S) {
#pragma unroll
      for (int i = 0; i < 2; ++i) {
        const int ch = tid + i * 128;
        const int r = ch >> 2, part = ch & 3;
        async_copy_b128(lds_addr(&bsm[(st + 1) & 1][r][part * 8]),
                        wT + (long)(colbase + r) * 512 + kk + 32 + part * 8);
      }
    }
    __builtin_prefetch(arow + kk + 64, 0, 1);
    v16bf a0 = load_a_bf16(arow + kk, 512);
    v16bf a1 = load_a_bf16(arow + (long)16 * 512 + kk, 512);
    const __bf16* bt = &bsm[st & 1][0][0];
#pragma unroll
    for (int cb = 0; cb < 4; ++cb) {
      v16bf bb = load_b_bf16t(bt + cb * 16 * 32, 32);  // ds_load_b128
      acc[0][cb] = wmma_bf16(a0, bb, acc[0][cb]);
      acc[1][cb] = wmma_bf16(a1, bb, acc[1][cb]);
    }
    wait_async_all();
    __syncthreads();
  }

  const int hi = (lane >> 4) << 3;
  const int nc = lane & 15;
#pragma unroll
  for (int cb = 0; cb < 4; ++cb) {
    const float bv = bias[colbase + cb * 16 + nc];
#pragma unroll
    for (int ra = 0; ra < 2; ++ra)
#pragma unroll
      for (int r = 0; r < 8; ++r) acc[ra][cb][r] += bv;
  }

  if (t < 2) {  // block-uniform branch
    const float mul = (t == 0) ? 0.125f : 1.0f;  // q gets d^-0.5
#pragma unroll
    for (int ra = 0; ra < 2; ++ra)
#pragma unroll
      for (int r = 0; r < 8; ++r) {
        float ss = 0.0f;
#pragma unroll
        for (int cb = 0; cb < 4; ++cb) ss += acc[ra][cb][r] * acc[ra][cb][r];
        ss = hsum16(ss);
        const float rn = rsqrtf(fmaxf(ss, 1e-12f)) * mul;
#pragma unroll
        for (int cb = 0; cb < 4; ++cb) acc[ra][cb][r] *= rn;
      }
    // transpose 32(n) x 64(d) tile through LDS -> [d][n]
#pragma unroll
    for (int ra = 0; ra < 2; ++ra)
#pragma unroll
      for (int r = 0; r < 8; ++r)
#pragma unroll
        for (int cb = 0; cb < 4; ++cb)
          tsm[wave][cb * 16 + nc][ra * 16 + r + hi] = (__bf16)acc[ra][cb][r];
    __syncthreads();
    __bf16* dst = (t == 0 ? qT : kT) + (long)bh * 64 * 4096 + nloc;
#pragma unroll
    for (int i = 0; i < 2; ++i) {
      const int d = i * 32 + lane;
      const uint4* lp = (const uint4*)&tsm[wave][d][0];
      uint4 v0 = lp[0], v1 = lp[1], v2 = lp[2], v3 = lp[3];
      uint4* gp = (uint4*)(dst + (long)d * 4096);
      gp[0] = v0; gp[1] = v1; gp[2] = v2; gp[3] = v3;
    }
  } else {
    __bf16* dst = vo + ((long)bh * 4096 + nloc) * 64;
#pragma unroll
    for (int ra = 0; ra < 2; ++ra)
#pragma unroll
      for (int r = 0; r < 8; ++r)
#pragma unroll
        for (int cb = 0; cb < 4; ++cb)
          dst[(ra * 16 + r + hi) * 64 + cb * 16 + nc] = (__bf16)acc[ra][cb][r];
  }
}

// ---------------------------------------------------------------------------
// Kernel 2: attn[b,h] = softmax_e( sum_n q[n,d] k[n,e] )  (64x64, K=4096)
// Output stored transposed attnT[e][d] via LDS.
// ---------------------------------------------------------------------------
__global__ __launch_bounds__(128) void attn_softmax_kernel(
    const __bf16* __restrict__ qT, const __bf16* __restrict__ kT,
    __bf16* __restrict__ attnT) {
  __shared__ __align__(16) __bf16 lds[64][72];
  const int lane = threadIdx.x & 31;
  const int wave = threadIdx.x >> 5;
  const int bh = blockIdx.x;
  const __bf16* qb = qT + (long)bh * 64 * 4096;
  const __bf16* kb = kT + (long)bh * 64 * 4096;
  const int d0 = wave * 16;

  v8f acc[4];
#pragma unroll
  for (int c = 0; c < 4; ++c) acc[c] = zero_v8f();

  for (int n0 = 0; n0 < 4096; n0 += 32) {
    __builtin_prefetch(qb + (long)d0 * 4096 + n0 + 64, 0, 1);
    v16bf a = load_a_bf16(qb + (long)d0 * 4096 + n0, 4096);
#pragma unroll
    for (int cb = 0; cb < 4; ++cb) {
      v16bf bb = load_b_bf16t(kb + (long)(cb * 16) * 4096 + n0, 4096);
      acc[cb] = wmma_bf16(a, bb, acc[cb]);
    }
  }
  const int hi = (lane >> 4) << 3;
  const int nc = lane & 15;
#pragma unroll
  for (int r = 0; r < 8; ++r) {
    float mx = -3.4e38f;
#pragma unroll
    for (int cb = 0; cb < 4; ++cb) mx = fmaxf(mx, acc[cb][r]);
    mx = hmax16(mx);
    float ss = 0.0f;
#pragma unroll
    for (int cb = 0; cb < 4; ++cb) {
      const float e = __expf(acc[cb][r] - mx);
      acc[cb][r] = e;
      ss += e;
    }
    ss = hsum16(ss);
    const float inv = 1.0f / ss;
#pragma unroll
    for (int cb = 0; cb < 4; ++cb) acc[cb][r] *= inv;
  }
#pragma unroll
  for (int r = 0; r < 8; ++r)
#pragma unroll
    for (int cb = 0; cb < 4; ++cb)
      lds[cb * 16 + nc][d0 + r + hi] = (__bf16)acc[cb][r];
  __syncthreads();
  const int e = threadIdx.x >> 1;
  const int half = threadIdx.x & 1;
  const uint4* lp = (const uint4*)&lds[e][half * 32];
  __bf16* dst = attnT + (long)bh * 4096 + e * 64 + half * 32;
  uint4 a0 = lp[0], a1 = lp[1], a2 = lp[2], a3 = lp[3];
  ((uint4*)dst)[0] = a0;
  ((uint4*)dst)[1] = a1;
  ((uint4*)dst)[2] = a2;
  ((uint4*)dst)[3] = a3;
}

// ---------------------------------------------------------------------------
// Kernel 3: out[n,e] = sum_d v[n,d] attnT[e,d]; store into [b][n][h*64+e]
// ---------------------------------------------------------------------------
__global__ __launch_bounds__(128) void av_kernel(
    const __bf16* __restrict__ v, const __bf16* __restrict__ attnT,
    __bf16* __restrict__ outp) {
  const int lane = threadIdx.x & 31;
  const int wave = threadIdx.x >> 5;
  const int nt = blockIdx.x;
  const int bh = blockIdx.y;
  const int b = bh >> 3, h = bh & 7;
  const int n0 = nt * 64 + wave * 16;
  const __bf16* vb = v + ((long)bh * 4096 + n0) * 64;
  const __bf16* ab = attnT + (long)bh * 4096;

  v8f acc[4];
#pragma unroll
  for (int c = 0; c < 4; ++c) acc[c] = zero_v8f();
#pragma unroll
  for (int kk = 0; kk < 64; kk += 32) {
    v16bf a = load_a_bf16(vb + kk, 64);
#pragma unroll
    for (int cb = 0; cb < 4; ++cb) {
      v16bf bb = load_b_bf16t(ab + (long)(cb * 16) * 64 + kk, 64);
      acc[cb] = wmma_bf16(a, bb, acc[cb]);
    }
  }
  const int hi = (lane >> 4) << 3;
  const int nc = lane & 15;
#pragma unroll
  for (int r = 0; r < 8; ++r) {
    const int n = n0 + r + hi;
    __bf16* dst = outp + ((long)b * 4096 + n) * 512 + h * 64;
#pragma unroll
    for (int cb = 0; cb < 4; ++cb) dst[cb * 16 + nc] = (__bf16)acc[cb][r];
  }
}

// ---------------------------------------------------------------------------
// Kernel 4: final = out_pre @ w_proj + b_proj.  Wave = 32x64 tile;
// B^T tile double-buffered in LDS via async copies.
// ---------------------------------------------------------------------------
__global__ __launch_bounds__(128) void proj_kernel(
    const __bf16* __restrict__ a_in, const __bf16* __restrict__ wT,
    const float* __restrict__ bias, float* __restrict__ out) {
  __shared__ __align__(16) __bf16 bsm[2][64][32];
  const int tid  = threadIdx.x;
  const int lane = tid & 31;
  const int wave = tid >> 5;
  const int rt = blockIdx.x;   // 0..255 (128-row tiles)
  const int ct = blockIdx.y;   // 0..7
  const int row0 = rt * 128 + wave * 32;
  const int col0 = ct * 64;

  v8f acc[2][4];
#pragma unroll
  for (int ra = 0; ra < 2; ++ra)
#pragma unroll
    for (int c = 0; c < 4; ++c) acc[ra][c] = zero_v8f();

  const __bf16* arow = a_in + (long)row0 * 512;
  const int S = 16;
  {
#pragma unroll
    for (int i = 0; i < 2; ++i) {
      const int ch = tid + i * 128;
      const int r = ch >> 2, part = ch & 3;
      async_copy_b128(lds_addr(&bsm[0][r][part * 8]),
                      wT + (long)(col0 + r) * 512 + part * 8);
    }
    wait_async_all();
    __syncthreads();
  }
  for (int st = 0; st < S; ++st) {
    const int kk = st * 32;
    if (st + 1 < S) {
#pragma unroll
      for (int i = 0; i < 2; ++i) {
        const int ch = tid + i * 128;
        const int r = ch >> 2, part = ch & 3;
        async_copy_b128(lds_addr(&bsm[(st + 1) & 1][r][part * 8]),
                        wT + (long)(col0 + r) * 512 + kk + 32 + part * 8);
      }
    }
    __builtin_prefetch(arow + kk + 64, 0, 1);
    v16bf a0 = load_a_bf16(arow + kk, 512);
    v16bf a1 = load_a_bf16(arow + (long)16 * 512 + kk, 512);
    const __bf16* bt = &bsm[st & 1][0][0];
#pragma unroll
    for (int cb = 0; cb < 4; ++cb) {
      v16bf bb = load_b_bf16t(bt + cb * 16 * 32, 32);
      acc[0][cb] = wmma_bf16(a0, bb, acc[0][cb]);
      acc[1][cb] = wmma_bf16(a1, bb, acc[1][cb]);
    }
    wait_async_all();
    __syncthreads();
  }
  const int hi = (lane >> 4) << 3;
  const int nc = lane & 15;
#pragma unroll
  for (int cb = 0; cb < 4; ++cb) {
    const float bv = bias[col0 + cb * 16 + nc];
#pragma unroll
    for (int ra = 0; ra < 2; ++ra)
#pragma unroll
      for (int r = 0; r < 8; ++r)
        out[((long)row0 + ra * 16 + r + hi) * 512 + col0 + cb * 16 + nc] =
            acc[ra][cb][r] + bv;
  }
}

// ---------------------------------------------------------------------------
extern "C" void kernel_launch(void* const* d_in, const int* in_sizes, int n_in,
                              void* d_out, int out_size, void* d_ws, size_t ws_size,
                              hipStream_t stream) {
  (void)in_sizes; (void)n_in; (void)out_size; (void)ws_size;
  const float* x      = (const float*)d_in[0];
  const float* w_qkv  = (const float*)d_in[1];
  const float* b_qkv  = (const float*)d_in[2];
  const float* w_proj = (const float*)d_in[3];
  const float* b_proj = (const float*)d_in[4];
  float* out = (float*)d_out;

  const size_t MB = 1024 * 1024;
  char* ws = (char*)d_ws;
  __bf16* xb     = (__bf16*)(ws);                  // 32 MB   [32768][512]
  __bf16* wqkvT  = (__bf16*)(ws + 32 * MB);        // 1.5 MB  [1536][512]
  __bf16* wprojT = (__bf16*)(ws + 34 * MB);        // 0.5 MB  [512][512]
  __bf16* qT     = (__bf16*)(ws + 35 * MB);        // 32 MB   [b][h][64][4096]
  __bf16* kT     = (__bf16*)(ws + 67 * MB);        // 32 MB
  __bf16* v      = (__bf16*)(ws + 99 * MB);        // 32 MB   [b][h][4096][64]
  __bf16* attnT  = (__bf16*)(ws + 131 * MB);       // 0.5 MB  [b][h][64][64]
  __bf16* outp   = (__bf16*)(ws + 132 * MB);       // 32 MB   [b][n][512]

  cast_bf16_kernel<<<8192, 256, 0, stream>>>(x, xb);
  transpose_cast_kernel<<<3072, 256, 0, stream>>>(w_qkv, wqkvT, 512, 1536);
  transpose_cast_kernel<<<1024, 256, 0, stream>>>(w_proj, wprojT, 512, 512);

  qkv_norm_kernel<<<dim3(256, 24), 128, 0, stream>>>(xb, wqkvT, b_qkv, qT, kT, v);
  attn_softmax_kernel<<<64, 128, 0, stream>>>(qT, kT, attnT);
  av_kernel<<<dim3(64, 64), 128, 0, stream>>>(v, attnT, outp);
  proj_kernel<<<dim3(256, 8), 128, 0, stream>>>(outp, wprojT, b_proj, out);
}
